// NL_56495999811923
// MI455X (gfx1250) — matrix operand order repaired and checked by reading, not compiled, not run on hardware
//
#include <hip/hip_runtime.h>
#include <math.h>

// ---------------------------------------------------------------------------
// CDNA5 (gfx1250) wave32 WMMA types & helpers
// ---------------------------------------------------------------------------
typedef __attribute__((ext_vector_type(16))) __bf16 v16bf;
typedef __attribute__((ext_vector_type(8)))  __bf16 v8bf;
typedef __attribute__((ext_vector_type(8)))  float  v8f;

#define BN_EPS 1e-5f

__device__ __forceinline__ __bf16 f2bf(float f) {
  union { float f; unsigned u; } v; v.f = f;
  unsigned r = v.u + 0x7fffu + ((v.u >> 16) & 1u);   // round-to-nearest-even
  union { unsigned short s; __bf16 b; } o; o.s = (unsigned short)(r >> 16);
  return o.b;
}

__device__ __forceinline__ v8f vzero8() {
  v8f z;
#pragma unroll
  for (int i = 0; i < 8; ++i) z[i] = 0.f;
  return z;
}

__device__ __forceinline__ v16bf bzero16() {
  v16bf z;
#pragma unroll
  for (int i = 0; i < 16; ++i) z[i] = f2bf(0.f);
  return z;
}

// A-fragment (16x32 bf16, M x K): lane holds row M=lane%16; element j maps to
// K = half*16 + j, half = lane/16.  `rowk` points at (row, kbase).
__device__ __forceinline__ v16bf load_afrag(const __bf16* rowk, int half) {
  const __bf16* p = rowk + half * 16;
  v8bf lo = *(const v8bf*)p;          // K = kbase + half*16 + 0..7
  v8bf hi = *(const v8bf*)(p + 8);    // K = kbase + half*16 + 8..15
  v16bf a;
#pragma unroll
  for (int i = 0; i < 8; ++i) { a[i] = lo[i]; a[i + 8] = hi[i]; }
  return a;
}

// B-fragment (32x16 bf16, K x N): lane holds col N=lane%16; element j maps to
// K = half*16 + j  -> 16 contiguous K values = one 32B load.
__device__ __forceinline__ v8f wmma_bf16(v16bf a, v16bf b, v8f c) {
  return __builtin_amdgcn_wmma_f32_16x16x32_bf16(false, a, false, b, (short)0, c,
                                                 false, false);
}

// jax.image.resize 'bilinear' (half-pixel centers, edge clamp)
__device__ __forceinline__ float bisample(const float* __restrict__ src, int Ct,
                                          int S, float inv, int bb, int c, int h,
                                          int w) {
  float sy = (h + 0.5f) * inv - 0.5f;
  float sx = (w + 0.5f) * inv - 0.5f;
  int y0 = (int)floorf(sy), x0i = (int)floorf(sx);
  float fy = sy - (float)y0, fx = sx - (float)x0i;
  int y1 = y0 + 1, x1i = x0i + 1;
  y0 = y0 < 0 ? 0 : (y0 > S - 1 ? S - 1 : y0);
  y1 = y1 < 0 ? 0 : (y1 > S - 1 ? S - 1 : y1);
  x0i = x0i < 0 ? 0 : (x0i > S - 1 ? S - 1 : x0i);
  x1i = x1i < 0 ? 0 : (x1i > S - 1 ? S - 1 : x1i);
  const float* p = src + ((size_t)bb * Ct + c) * S * S;
  float v00 = p[y0 * S + x0i], v01 = p[y0 * S + x1i];
  float v10 = p[y1 * S + x0i], v11 = p[y1 * S + x1i];
  float a = v00 + fx * (v01 - v00);
  float b = v10 + fx * (v11 - v10);
  return a + fy * (b - a);
}

// ---------------------------------------------------------------------------
// K1: upsample+concat (on the fly, in LDS) -> conv1x1 720->48 -> BN -> ReLU
// out: x stored pixel-major [B][N][48] as bf16 (for later GEMMs) and fp32
// (residual). One workgroup per (b, image row h): 128 pixels, 8 waves.
// ---------------------------------------------------------------------------
__global__ __launch_bounds__(256) void k1_fuse(
    const float* __restrict__ x0p, const float* __restrict__ x1p,
    const float* __restrict__ x2p, const float* __restrict__ x3p,
    const float* __restrict__ fw, const float* __restrict__ fb,
    const float* __restrict__ fg, const float* __restrict__ fbeta,
    const float* __restrict__ fmean, const float* __restrict__ fvar,
    __bf16* __restrict__ xb, float* __restrict__ xf) {
  __shared__ __align__(32) unsigned char smem[34176];
  __bf16* Xs = (__bf16*)smem;             // [128 px][96 K], stride 96 (192B)
  __bf16* Ws = (__bf16*)(smem + 24576);   // [48 ch][96 K]
  float* scl = (float*)(smem + 33792);    // [48]
  float* sft = (float*)(smem + 33984);    // [48]

  const int tid = threadIdx.x;
  const int b = blockIdx.x >> 7;
  const int h = blockIdx.x & 127;

  if (tid < 48) {
    float s = fg[tid] * rsqrtf(fvar[tid] + BN_EPS);
    scl[tid] = s;
    sft[tid] = fb[tid] * s + fbeta[tid] - fmean[tid] * s;
  }

  const int wv = tid >> 5, lane = tid & 31, half = lane >> 4, ln = lane & 15;
  const int pxN = wv * 16 + ln;  // this lane's output pixel column

  v8f acc[3];
#pragma unroll
  for (int m = 0; m < 3; ++m) acc[m] = vzero8();

  const float* srcs[4] = {x0p, x1p, x2p, x3p};
  const float invs[4] = {1.f, 0.5f, 0.25f, 0.125f};
  const int Ss[4] = {128, 64, 32, 16};
  const int Cts[4] = {48, 96, 192, 384};
  // K (=736 padded) split into 8 chunks of <=96
  const int cSrc[8] = {0, 1, 2, 2, 3, 3, 3, 3};
  const int cChB[8] = {0, 0, 0, 96, 0, 96, 192, 288};
  const int cKc[8]  = {48, 96, 96, 96, 96, 96, 96, 96};
  const int cKp[8]  = {64, 96, 96, 96, 96, 96, 96, 96};
  const int cWb[8]  = {0, 48, 144, 240, 336, 432, 528, 624};

#pragma unroll
  for (int ck = 0; ck < 8; ++ck) {
    const int sI = cSrc[ck], chB = cChB[ck], Kc = cKc[ck], Kp = cKp[ck];
    const int wb = cWb[ck];
    __syncthreads();
    // stage upsampled concat chunk, K-contiguous per pixel
    for (int i = tid; i < (Kp << 7); i += 256) {
      int k = i >> 7, px = i & 127;
      float v = 0.f;
      if (k < Kc) {
        if (sI == 0)
          v = x0p[((size_t)(b * 48 + k) * 128 + h) * 128 + px];
        else
          v = bisample(srcs[sI], Cts[sI], Ss[sI], invs[sI], b, chB + k, h, px);
      }
      Xs[px * 96 + k] = f2bf(v);
    }
    // stage weight chunk (zero K-pad)
    for (int i = tid; i < 48 * Kp; i += 256) {
      int k = i % Kp, o = i / Kp;
      Ws[o * 96 + k] = f2bf((k < Kc) ? fw[o * 720 + wb + k] : 0.f);
    }
    __syncthreads();
    const int nks = Kp >> 5;
    for (int kk = 0; kk < nks; ++kk) {
      v16bf bf = *(const v16bf*)&Xs[pxN * 96 + kk * 32 + half * 16];
#pragma unroll
      for (int m = 0; m < 3; ++m) {
        v16bf af = load_afrag(&Ws[(m * 16 + ln) * 96 + kk * 32], half);
        acc[m] = wmma_bf16(af, bf, acc[m]);
      }
    }
  }

  // epilogue: bias+BN+ReLU, store x pixel-major (bf16 + fp32)
  const int n = h * 128 + pxN;
  const size_t nb = ((size_t)b * 16384 + n) * 48;
#pragma unroll
  for (int m = 0; m < 3; ++m) {
    const int ch0 = m * 16 + half * 8;
    float vs[8];
    v8bf pk;
#pragma unroll
    for (int r = 0; r < 8; ++r) {
      float v = acc[m][r] * scl[ch0 + r] + sft[ch0 + r];
      v = fmaxf(v, 0.f);
      vs[r] = v;
      pk[r] = f2bf(v);
    }
    *(v8bf*)(xb + nb + ch0) = pk;
    *(float4*)(xf + nb + ch0) = make_float4(vs[0], vs[1], vs[2], vs[3]);
    *(float4*)(xf + nb + ch0 + 4) = make_float4(vs[4], vs[5], vs[6], vs[7]);
  }
}

// ---------------------------------------------------------------------------
// K2: phi = BN(Wp x), g = BN(Wg x) (WMMA, K=48 pad 64), then per-tile partial
// M[c][d] = sum_px phi[c,px] g[d,px] via WMMA (K = 256 px, 32 px per wave).
// One workgroup per (b, 256-pixel tile).
// ---------------------------------------------------------------------------
__global__ __launch_bounds__(256) void k2_gphi(
    const __bf16* __restrict__ xb, const float* __restrict__ gw,
    const float* __restrict__ gb, const float* __restrict__ gg,
    const float* __restrict__ gbe, const float* __restrict__ gme,
    const float* __restrict__ gva, const float* __restrict__ pw,
    const float* __restrict__ pb, const float* __restrict__ pg,
    const float* __restrict__ pbe, const float* __restrict__ pme,
    const float* __restrict__ pva, float* __restrict__ part) {
  __shared__ __align__(32) unsigned char smem[62208];
  __bf16* Wp = (__bf16*)smem;                // [48][64] phi weights
  __bf16* Wg = (__bf16*)(smem + 6144);       // [48][64] g weights
  float* Ml = (float*)smem;                  // [48][48] (aliases W, phase B)
  __bf16* phiL = (__bf16*)(smem + 12288);    // [48 ch][256 px]
  __bf16* gL = (__bf16*)(smem + 36864);      // [48 ch][256 px]
  float* tab = (float*)(smem + 61440);       // sP[48] hP[48] sG[48] hG[48]

  const int tid = threadIdx.x;
  const int b = blockIdx.x >> 6;
  const int px0 = (blockIdx.x & 63) * 256;

  for (int i = tid; i < 48 * 64; i += 256) {
    int o = i >> 6, k = i & 63;
    Wp[i] = f2bf((k < 48) ? pw[o * 48 + k] : 0.f);
    Wg[i] = f2bf((k < 48) ? gw[o * 48 + k] : 0.f);
  }
  if (tid < 48) {
    float sp = pg[tid] * rsqrtf(pva[tid] + BN_EPS);
    tab[tid] = sp;
    tab[48 + tid] = pb[tid] * sp + pbe[tid] - pme[tid] * sp;
    float sg = gg[tid] * rsqrtf(gva[tid] + BN_EPS);
    tab[96 + tid] = sg;
    tab[144 + tid] = gb[tid] * sg + gbe[tid] - gme[tid] * sg;
  }
  __syncthreads();

  const int wv = tid >> 5, lane = tid & 31, half = lane >> 4, ln = lane & 15;

  // phase A: phi & g GEMMs; B-fragments straight from global (pixel-major x)
  v8f aP[3][2], aG[3][2];
#pragma unroll
  for (int m = 0; m < 3; ++m)
#pragma unroll
    for (int nt = 0; nt < 2; ++nt) { aP[m][nt] = vzero8(); aG[m][nt] = vzero8(); }

#pragma unroll
  for (int kk = 0; kk < 2; ++kk) {
    v16bf bf[2];
    const int ch0 = kk * 32 + half * 16;
#pragma unroll
    for (int nt = 0; nt < 2; ++nt) {
      const int col = wv * 32 + nt * 16 + ln;
      if (ch0 < 48)
        bf[nt] = *(const v16bf*)(xb + ((size_t)b * 16384 + px0 + col) * 48 + ch0);
      else
        bf[nt] = bzero16();
    }
#pragma unroll
    for (int m = 0; m < 3; ++m) {
      v16bf afP = load_afrag(&Wp[(m * 16 + ln) * 64 + kk * 32], half);
      v16bf afG = load_afrag(&Wg[(m * 16 + ln) * 64 + kk * 32], half);
#pragma unroll
      for (int nt = 0; nt < 2; ++nt) {
        aP[m][nt] = wmma_bf16(afP, bf[nt], aP[m][nt]);
        aG[m][nt] = wmma_bf16(afG, bf[nt], aG[m][nt]);
      }
    }
  }

  // BN, transpose through LDS into [ch][px] (px-contiguous = WMMA K stripes)
#pragma unroll
  for (int m = 0; m < 3; ++m)
#pragma unroll
    for (int nt = 0; nt < 2; ++nt) {
      const int col = wv * 32 + nt * 16 + ln;
#pragma unroll
      for (int r = 0; r < 8; ++r) {
        const int ch = m * 16 + half * 8 + r;
        phiL[ch * 256 + col] = f2bf(aP[m][nt][r] * tab[ch] + tab[48 + ch]);
        gL[ch * 256 + col] = f2bf(aG[m][nt][r] * tab[96 + ch] + tab[144 + ch]);
      }
    }
  __syncthreads();

  for (int i = tid; i < 2304; i += 256) Ml[i] = 0.f;  // reuse dead W region
  __syncthreads();

  // phase B: per-wave 48x48 over its 32-px K slice (1 WMMA k-step, 9 tiles)
  const int koff = wv * 32;
  v16bf afPhi[3], bG[3];
#pragma unroll
  for (int mt = 0; mt < 3; ++mt)
    afPhi[mt] = load_afrag(&phiL[(mt * 16 + ln) * 256 + koff], half);
#pragma unroll
  for (int nt = 0; nt < 3; ++nt)
    bG[nt] = *(const v16bf*)&gL[(nt * 16 + ln) * 256 + koff + half * 16];

  v8f mm[3][3];
#pragma unroll
  for (int mt = 0; mt < 3; ++mt)
#pragma unroll
    for (int nt = 0; nt < 3; ++nt)
      mm[mt][nt] = wmma_bf16(afPhi[mt], bG[nt], vzero8());

  // deterministic round-robin cross-wave reduction into Ml
  for (int round = 0; round < 8; ++round) {
    if (wv == round) {
#pragma unroll
      for (int mt = 0; mt < 3; ++mt)
#pragma unroll
        for (int nt = 0; nt < 3; ++nt)
#pragma unroll
          for (int r = 0; r < 8; ++r) {
            const int c = mt * 16 + half * 8 + r;
            const int d = nt * 16 + ln;
            Ml[c * 48 + d] += mm[mt][nt][r];
          }
    }
    __syncthreads();
  }
  for (int i = tid; i < 2304; i += 256)
    part[(size_t)blockIdx.x * 2304 + i] = Ml[i];
}

// ---------------------------------------------------------------------------
// K2b: reduce 64 per-tile partials -> M[b][c][d] / 16384
// ---------------------------------------------------------------------------
__global__ __launch_bounds__(256) void k2b_reduce(const float* __restrict__ part,
                                                  float* __restrict__ Mg) {
  int gid = blockIdx.x * 256 + threadIdx.x;
  if (gid >= 8 * 2304) return;
  int b = gid / 2304, e = gid % 2304;
  const float* p = part + (size_t)b * 64 * 2304 + e;
  float s = 0.f;
  for (int c = 0; c < 64; ++c) s += p[(size_t)c * 2304];
  Mg[gid] = s * (1.0f / 16384.0f);
}

// ---------------------------------------------------------------------------
// K3: theta = BN(Wt x) (WMMA) -> out = M^T * theta + x, channel-major output.
// ---------------------------------------------------------------------------
__global__ __launch_bounds__(256) void k3_out(
    const __bf16* __restrict__ xb, const float* __restrict__ xf,
    const float* __restrict__ tw, const float* __restrict__ tb,
    const float* __restrict__ tg, const float* __restrict__ tbe,
    const float* __restrict__ tme, const float* __restrict__ tva,
    const float* __restrict__ Mg, float* __restrict__ out) {
  __shared__ __align__(32) unsigned char smem[45440];
  __bf16* Wt = (__bf16*)smem;              // [48][64] theta weights
  __bf16* Mt = (__bf16*)(smem + 6144);     // [48 d][64 c]  Mt[d][c]=M[c][d]/N
  __bf16* thL = (__bf16*)(smem + 12288);   // [256 px][64 ch]
  float* tab = (float*)(smem + 45056);     // s[48] h[48]

  const int tid = threadIdx.x;
  const int b = blockIdx.x >> 6;
  const int px0 = (blockIdx.x & 63) * 256;

  for (int i = tid; i < 48 * 64; i += 256) {
    int o = i >> 6, k = i & 63;
    Wt[i] = f2bf((k < 48) ? tw[o * 48 + k] : 0.f);
    Mt[i] = f2bf((k < 48) ? Mg[(size_t)b * 2304 + k * 48 + o] : 0.f);
  }
  for (int i = tid; i < 256 * 16; i += 256) {  // zero channel pad 48..63
    int px = i >> 4, j = i & 15;
    thL[px * 64 + 48 + j] = f2bf(0.f);
  }
  if (tid < 48) {
    float s = tg[tid] * rsqrtf(tva[tid] + BN_EPS);
    tab[tid] = s;
    tab[48 + tid] = tb[tid] * s + tbe[tid] - tme[tid] * s;
  }
  __syncthreads();

  const int wv = tid >> 5, lane = tid & 31, half = lane >> 4, ln = lane & 15;

  // phase A: theta GEMM
  v8f aT[3][2];
#pragma unroll
  for (int m = 0; m < 3; ++m)
#pragma unroll
    for (int nt = 0; nt < 2; ++nt) aT[m][nt] = vzero8();

#pragma unroll
  for (int kk = 0; kk < 2; ++kk) {
    v16bf bf[2];
    const int ch0 = kk * 32 + half * 16;
#pragma unroll
    for (int nt = 0; nt < 2; ++nt) {
      const int col = wv * 32 + nt * 16 + ln;
      if (ch0 < 48)
        bf[nt] = *(const v16bf*)(xb + ((size_t)b * 16384 + px0 + col) * 48 + ch0);
      else
        bf[nt] = bzero16();
    }
#pragma unroll
    for (int m = 0; m < 3; ++m) {
      v16bf af = load_afrag(&Wt[(m * 16 + ln) * 64 + kk * 32], half);
#pragma unroll
      for (int nt = 0; nt < 2; ++nt) aT[m][nt] = wmma_bf16(af, bf[nt], aT[m][nt]);
    }
  }

  // BN -> thL[px][ch] (ch-contiguous, packed 16B stores)
#pragma unroll
  for (int m = 0; m < 3; ++m)
#pragma unroll
    for (int nt = 0; nt < 2; ++nt) {
      const int col = wv * 32 + nt * 16 + ln;
      const int ch0 = m * 16 + half * 8;
      v8bf pk;
#pragma unroll
      for (int r = 0; r < 8; ++r)
        pk[r] = f2bf(aT[m][nt][r] * tab[ch0 + r] + tab[48 + ch0 + r]);
      *(v8bf*)&thL[col * 64 + ch0] = pk;
    }
  __syncthreads();

  // phase B: out[d,px] = Mt[d][c] * th[c,px]  (K = 64 incl. zero pad)
  v8f aO[3][2];
#pragma unroll
  for (int mt = 0; mt < 3; ++mt)
#pragma unroll
    for (int nt = 0; nt < 2; ++nt) aO[mt][nt] = vzero8();

#pragma unroll
  for (int kk = 0; kk < 2; ++kk) {
    v16bf bf2[2];
#pragma unroll
    for (int nt = 0; nt < 2; ++nt)
      bf2[nt] = *(const v16bf*)&thL[(wv * 32 + nt * 16 + ln) * 64 + kk * 32 +
                                    half * 16];
#pragma unroll
    for (int mt = 0; mt < 3; ++mt) {
      v16bf am = load_afrag(&Mt[(mt * 16 + ln) * 64 + kk * 32], half);
#pragma unroll
      for (int nt = 0; nt < 2; ++nt) aO[mt][nt] = wmma_bf16(am, bf2[nt], aO[mt][nt]);
    }
  }

  // residual add (contiguous fp32 loads) + channel-major scatter store
#pragma unroll
  for (int mt = 0; mt < 3; ++mt)
#pragma unroll
    for (int nt = 0; nt < 2; ++nt) {
      const int d0 = mt * 16 + half * 8;
      const int n = px0 + wv * 32 + nt * 16 + ln;
      const float* xr = xf + ((size_t)b * 16384 + n) * 48 + d0;
      float4 xa = *(const float4*)xr;
      float4 xc = *(const float4*)(xr + 4);
      float xv[8] = {xa.x, xa.y, xa.z, xa.w, xc.x, xc.y, xc.z, xc.w};
#pragma unroll
      for (int r = 0; r < 8; ++r)
        out[((size_t)(b * 48 + d0 + r)) * 16384 + n] = aO[mt][nt][r] + xv[r];
    }
}

// ---------------------------------------------------------------------------
extern "C" void kernel_launch(void* const* d_in, const int* in_sizes, int n_in,
                              void* d_out, int out_size, void* d_ws,
                              size_t ws_size, hipStream_t stream) {
  (void)in_sizes; (void)n_in; (void)out_size; (void)ws_size;
  const float* x0 = (const float*)d_in[0];
  const float* x1 = (const float*)d_in[1];
  const float* x2 = (const float*)d_in[2];
  const float* x3 = (const float*)d_in[3];
  const float* fw = (const float*)d_in[4];
  const float* fb = (const float*)d_in[5];
  const float* fg = (const float*)d_in[6];
  const float* fbe = (const float*)d_in[7];
  const float* fme = (const float*)d_in[8];
  const float* fva = (const float*)d_in[9];
  const float* gw = (const float*)d_in[10];
  const float* gb = (const float*)d_in[11];
  const float* gg = (const float*)d_in[12];
  const float* gbe = (const float*)d_in[13];
  const float* gme = (const float*)d_in[14];
  const float* gva = (const float*)d_in[15];
  const float* tw = (const float*)d_in[16];
  const float* tb = (const float*)d_in[17];
  const float* tg = (const float*)d_in[18];
  const float* tbe = (const float*)d_in[19];
  const float* tme = (const float*)d_in[20];
  const float* tva = (const float*)d_in[21];
  const float* pw = (const float*)d_in[22];
  const float* pb = (const float*)d_in[23];
  const float* pg = (const float*)d_in[24];
  const float* pbe = (const float*)d_in[25];
  const float* pme = (const float*)d_in[26];
  const float* pva = (const float*)d_in[27];

  // workspace layout (bytes): x bf16 [B][N][48] | x fp32 [B][N][48]
  //                           | partial M [512][2304] | M [8][2304]
  char* ws = (char*)d_ws;
  __bf16* xb = (__bf16*)ws;                       // 12,582,912 B
  float* xf = (float*)(ws + 12582912);            // 25,165,824 B
  float* part = (float*)(ws + 37748736);          //  4,718,592 B
  float* Mg = (float*)(ws + 42467328);            //     73,728 B

  k1_fuse<<<1024, 256, 0, stream>>>(x0, x1, x2, x3, fw, fb, fg, fbe, fme, fva,
                                    xb, xf);
  k2_gphi<<<512, 256, 0, stream>>>(xb, gw, gb, gg, gbe, gme, gva, pw, pb, pg,
                                   pbe, pme, pva, part);
  k2b_reduce<<<72, 256, 0, stream>>>(part, Mg);
  k3_out<<<512, 256, 0, stream>>>(xb, xf, tw, tb, tg, tbe, tme, tva, Mg,
                                  (float*)d_out);
}
